// NTM_72593537237207
// MI455X (gfx1250) — matrix operand order repaired and checked by reading, not compiled, not run on hardware
//
#include <hip/hip_runtime.h>
#include <math.h>

// ---- problem sizes ----
constexpr int Bsz = 256;     // batch
constexpr int Nn  = 2048;    // memory rows
constexpr int Md  = 64;      // memory width
constexpr int Cs  = 512;     // cell size
constexpr int In  = 256;     // input
constexpr int Od  = 256;     // output
constexpr int Kc  = In + Md + Cs;   // 832  concat K for gates GEMM
constexpr int Gt  = 4 * Cs;         // 2048 gate columns
constexpr int Ar  = Md + 6;         // 70
constexpr int Aw  = 3 * Md + 6;     // 198
constexpr int Ao  = Ar + Aw;        // 268

typedef __attribute__((ext_vector_type(16))) __bf16 v16bf;
typedef __attribute__((ext_vector_type(8)))  __bf16 v8bf;
typedef __attribute__((ext_vector_type(8)))  float  v8f;

typedef unsigned int u32x4 __attribute__((ext_vector_type(4)));
typedef int          i32x4 __attribute__((ext_vector_type(4)));
typedef int          i32x8 __attribute__((ext_vector_type(8)));
typedef __attribute__((address_space(3))) char lds_char;

__device__ __forceinline__ float sigf(float x)      { return 1.0f / (1.0f + __expf(-x)); }
__device__ __forceinline__ float softplusf(float x) { return (x > 20.0f) ? x : log1pf(__expf(x)); }

// ---------------- K0a: pack [x | prev_read | h] -> bf16 [Bsz x Kc] ----------------
__global__ void k_pack_inp(const float* __restrict__ x, const float* __restrict__ pr,
                           const float* __restrict__ h, __bf16* __restrict__ inp) {
    int i = blockIdx.x * blockDim.x + threadIdx.x;
    if (i >= Bsz * Kc) return;
    int b = i / Kc, c = i % Kc;
    float v;
    if (c < In)           v = x[b * In + c];
    else if (c < In + Md) v = pr[b * Md + (c - In)];
    else                  v = h[b * Cs + (c - In - Md)];
    inp[i] = (__bf16)v;
}

// ---------------- K0b: pack [W_ih | W_hh] -> bf16 [Gt x Kc] ----------------
__global__ void k_pack_w(const float* __restrict__ wih, const float* __restrict__ whh,
                         __bf16* __restrict__ w) {
    int i = blockIdx.x * blockDim.x + threadIdx.x;
    if (i >= Gt * Kc) return;
    int r = i / Kc, c = i % Kc;
    float v = (c < In + Md) ? wih[r * (In + Md) + c] : whh[r * Cs + (c - In - Md)];
    w[i] = (__bf16)v;
}

// ---------------- K1: gates = inp @ Wcat^T + b_ih + b_hh   (WMMA bf16) ----------------
// One 16x16 output tile per wave32; K walked in steps of 32 (26 iters).
__global__ void k_gates_wmma(const __bf16* __restrict__ A, const __bf16* __restrict__ W,
                             const float* __restrict__ bih, const float* __restrict__ bhh,
                             float* __restrict__ gates) {
    int lane = threadIdx.x & 31;
    int wave = blockIdx.x * (blockDim.x >> 5) + (threadIdx.x >> 5);
    int tn = wave & 127;   // 128 column tiles over Gt=2048
    int tm = wave >> 7;    // 16 row tiles over Bsz=256
    int m    = lane & 15;
    int half = lane >> 4;

    const __bf16* arow = A + (tm * 16 + m) * Kc;   // row of inp  (M = m)
    const __bf16* brow = W + (tn * 16 + m) * Kc;   // row of Wcat (N = m, same lane bits)

    v8f acc = {};
    for (int k0 = 0; k0 < Kc; k0 += 32) {
        v8bf  alo = *(const v8bf*)(arow + k0 + 8 * half);
        v8bf  ahi = *(const v8bf*)(arow + k0 + 16 + 8 * half);
        v8bf  blo = *(const v8bf*)(brow + k0 + 8 * half);
        v8bf  bhi = *(const v8bf*)(brow + k0 + 16 + 8 * half);
        v16bf af, bf;
#pragma unroll
        for (int i = 0; i < 8; ++i) { af[i] = alo[i]; af[i + 8] = ahi[i]; bf[i] = blo[i]; bf[i + 8] = bhi[i]; }
        acc = __builtin_amdgcn_wmma_f32_16x16x32_bf16(false, af, false, bf,
                                                      (short)0, acc, false, false);
    }
    int col = tn * 16 + m;
    float bias = bih[col] + bhh[col];
#pragma unroll
    for (int r = 0; r < 8; ++r) {
        int row = tm * 16 + r + 8 * half;       // C/D layout: VGPR r -> M = r + 8*half
        gates[row * Gt + col] = acc[r] + bias;
    }
}

// ---------------- K2: LSTM elementwise ----------------
__global__ void k_lstm(const float* __restrict__ gates, const float* __restrict__ c_in,
                       float* __restrict__ h_new, float* __restrict__ c_new) {
    int i = blockIdx.x * blockDim.x + threadIdx.x;
    if (i >= Bsz * Cs) return;
    int b = i / Cs, q = i % Cs;
    const float* g = gates + b * Gt;
    float ig = sigf(g[q]);
    float fg = sigf(g[q + Cs]);
    float gg = tanhf(g[q + 2 * Cs]);
    float og = sigf(g[q + 3 * Cs]);
    float c  = fg * c_in[i] + ig * gg;
    c_new[i] = c;
    h_new[i] = og * tanhf(c);
}

// ---------------- K3: [read_out | write_out] = h_new @ [W_r ; W_w]^T ----------------
__global__ void k_addr_gemm(const float* __restrict__ h,
                            const float* __restrict__ wr, const float* __restrict__ br,
                            const float* __restrict__ ww, const float* __restrict__ bw,
                            float* __restrict__ aout) {
    int i = blockIdx.x * blockDim.x + threadIdx.x;
    if (i >= Bsz * Ao) return;
    int b = i / Ao, j = i % Ao;
    const float* wrow; float bias;
    if (j < Ar) { wrow = wr + j * Cs;        bias = br[j]; }
    else        { wrow = ww + (j - Ar) * Cs; bias = bw[j - Ar]; }
    const float* hv = h + b * Cs;
    float acc = bias;
    for (int k = 0; k < Cs; ++k) acc += hv[k] * wrow[k];
    aout[i] = acc;
}

// ---------------- K4: per-batch head params + erase/add ----------------
// params[b*16 + ...]: 0:||k_r||+eps 1:beta_r 2:g_r 3..5:s_r 6:gamma_r
//                     7:||k_w||+eps 8:beta_w 9:g_w 10..12:s_w 13:gamma_w
__global__ void k_params(const float* __restrict__ aout, float* __restrict__ params,
                         float* __restrict__ eras, float* __restrict__ addv) {
    int b = blockIdx.x;
    int t = threadIdx.x;   // 64 threads
    __shared__ float red[64];
    const float* a = aout + b * Ao;

    float kr = a[t];
    red[t] = kr * kr; __syncthreads();
    for (int s = 32; s > 0; s >>= 1) { if (t < s) red[t] += red[t + s]; __syncthreads(); }
    if (t == 0) params[b * 16 + 0] = sqrtf(red[0]) + 1e-8f;
    __syncthreads();

    float kw = a[Ar + t];
    red[t] = kw * kw; __syncthreads();
    for (int s = 32; s > 0; s >>= 1) { if (t < s) red[t] += red[t + s]; __syncthreads(); }
    if (t == 0) params[b * 16 + 7] = sqrtf(red[0]) + 1e-8f;

    eras[b * Md + t] = sigf(a[2 * Ar + t]);
    addv[b * Md + t] = a[2 * Ar + Md + t];

    if (t == 0) {
        params[b * 16 + 1] = softplusf(a[Md]);
        params[b * 16 + 2] = sigf(a[Md + 1]);
        float s0 = a[Md + 2], s1 = a[Md + 3], s2 = a[Md + 4];
        float mx = fmaxf(s0, fmaxf(s1, s2));
        float e0 = __expf(s0 - mx), e1 = __expf(s1 - mx), e2 = __expf(s2 - mx);
        float S = e0 + e1 + e2;
        params[b * 16 + 3] = e0 / S; params[b * 16 + 4] = e1 / S; params[b * 16 + 5] = e2 / S;
        params[b * 16 + 6] = 1.0f + softplusf(a[Md + 5]);

        params[b * 16 + 8] = softplusf(a[Ar + Md]);
        params[b * 16 + 9] = sigf(a[Ar + Md + 1]);
        float t0 = a[Ar + Md + 2], t1 = a[Ar + Md + 3], t2 = a[Ar + Md + 4];
        float mw = fmaxf(t0, fmaxf(t1, t2));
        float f0 = __expf(t0 - mw), f1 = __expf(t1 - mw), f2 = __expf(t2 - mw);
        float T = f0 + f1 + f2;
        params[b * 16 + 10] = f0 / T; params[b * 16 + 11] = f1 / T; params[b * 16 + 12] = f2 / T;
        params[b * 16 + 13] = 1.0f + softplusf(a[Ar + Md + 5]);
    }
}

// ---------------- K5: streaming pass 1 — cosine sims via TDM-staged LDS tiles ----------------
// Each block stages SIM_ROWS x 64 f32 (32 KB) into LDS with one TENSOR_LOAD_TO_LDS
// (D# per ISA §8.3–8.6), waits on TENSORcnt, then 16 lanes/row reduce out of LDS.
constexpr int SIM_ROWS = 128;

__global__ void k_sim(const float* __restrict__ mem, const float* __restrict__ aout,
                      const float* __restrict__ params,
                      float* __restrict__ zr, float* __restrict__ zw) {
    __shared__ alignas(16) float tile[SIM_ROWS * Md];
    const long row0 = (long)blockIdx.x * SIM_ROWS;           // first global memory row
    const int  b    = (int)(row0 / Nn);                      // SIM_ROWS divides Nn

#if defined(__gfx1250__) && __has_builtin(__builtin_amdgcn_tensor_load_to_lds) && \
    __has_builtin(__builtin_amdgcn_s_wait_tensorcnt)
    if (threadIdx.x < 32) {   // wave 0 programs the Tensor Data Mover
        unsigned lds_off = (unsigned)(unsigned long long)(lds_char*)tile;
        unsigned long long ga = (unsigned long long)(const void*)(mem + row0 * Md);
        // D# group0: count=1, lds_addr, 57b global_addr, type=2 ("image")
        u32x4 g0 = { 1u, lds_off, (unsigned)ga,
                     (unsigned)((ga >> 32) & 0x01FFFFFFull) | 0x80000000u };
        // D# group1: data_size=4B(code 2); tensor_dim0=64, tensor_dim1=SIM_ROWS;
        //            tile_dim0=64, tile_dim1=SIM_ROWS; tensor_dim0_stride=64
        i32x8 g1 = { 0x00020000,          // workgroup_mask=0, data_size=2
                     64 << 16,            // tensor_dim0[15:0] in bits 63:48
                     SIM_ROWS << 16,      // tensor_dim0[31:16]=0 | tensor_dim1[15:0]
                     64 << 16,            // tensor_dim1[31:16]=0 | tile_dim0
                     SIM_ROWS,            // tile_dim1 | tile_dim2=0
                     64,                  // tensor_dim0_stride[31:0]
                     0, 0 };              // stride0 hi / tensor_dim1_stride (2D: unused)
        i32x4 gz = { 0, 0, 0, 0 };
#if __clang_major__ >= 23
        i32x8 gz8 = { 0, 0, 0, 0, 0, 0, 0, 0 };
        __builtin_amdgcn_tensor_load_to_lds(g0, g1, gz, gz, gz8, 0);
#else
        __builtin_amdgcn_tensor_load_to_lds(g0, g1, gz, gz, 0);
#endif
        __builtin_amdgcn_s_wait_tensorcnt(0);
    }
    __syncthreads();
#else
    // Fallback: cooperative staging with plain b128 loads
    for (int i = threadIdx.x; i < SIM_ROWS * Md / 4; i += blockDim.x) {
        float4 v = *(const float4*)(mem + row0 * Md + (long)i * 4);
        *(float4*)&tile[i * 4] = v;
    }
    __syncthreads();
#endif

    const float* a = aout + b * Ao;
    const int c    = threadIdx.x & 15;       // sixteenth-row lane (4 floats)
    const int rgrp = threadIdx.x >> 4;       // 16 row-groups per pass
    float kr0 = a[c * 4], kr1 = a[c * 4 + 1], kr2 = a[c * 4 + 2], kr3 = a[c * 4 + 3];
    float kw0 = a[Ar + c * 4], kw1 = a[Ar + c * 4 + 1],
          kw2 = a[Ar + c * 4 + 2], kw3 = a[Ar + c * 4 + 3];
    const float* p = params + b * 16;
    float nkr = p[0], betar = p[1], nkw = p[7], betaw = p[8];

#pragma unroll
    for (int ps = 0; ps < SIM_ROWS / 16; ++ps) {
        int rl = ps * 16 + rgrp;
        const float* mrow = &tile[rl * Md + c * 4];
        float m0 = mrow[0], m1 = mrow[1], m2 = mrow[2], m3 = mrow[3];
        float dr = m0 * kr0 + m1 * kr1 + m2 * kr2 + m3 * kr3;
        float dw = m0 * kw0 + m1 * kw1 + m2 * kw2 + m3 * kw3;
        float ss = m0 * m0 + m1 * m1 + m2 * m2 + m3 * m3;
#pragma unroll
        for (int off = 1; off < 16; off <<= 1) {
            dr += __shfl_xor(dr, off, 32);
            dw += __shfl_xor(dw, off, 32);
            ss += __shfl_xor(ss, off, 32);
        }
        if (c == 0) {
            float nm = sqrtf(ss) + 1e-8f;
            long row = row0 + rl;
            zr[row] = betar * dr / (nkr * nm);
            zw[row] = betaw * dw / (nkw * nm);
        }
    }
}

// ---------------- K6: softmax + interpolate + shift + sharpen + renorm ----------------
__global__ void k_address(const float* __restrict__ zr, const float* __restrict__ zw,
                          const float* __restrict__ read_w, const float* __restrict__ write_w,
                          const float* __restrict__ params,
                          float* __restrict__ w_read, float* __restrict__ w_write) {
    int b    = blockIdx.x >> 1;
    int head = blockIdx.x & 1;
    const float* z    = (head ? zw : zr) + (long)b * Nn;
    const float* prev = (head ? write_w : read_w) + (long)b * Nn;
    const float* p    = params + b * 16 + (head ? 7 : 0);
    float* wout       = (head ? w_write : w_read) + (long)b * Nn;
    float g = p[2], s0 = p[3], s1 = p[4], s2 = p[5], gamma = p[6];

    __shared__ float sw[Nn];
    __shared__ float red[256];
    int t = threadIdx.x;

    float zv[8];
    float lm = -1e30f;
#pragma unroll
    for (int i = 0; i < 8; ++i) { zv[i] = z[t + i * 256]; lm = fmaxf(lm, zv[i]); }
    red[t] = lm; __syncthreads();
    for (int s = 128; s > 0; s >>= 1) { if (t < s) red[t] = fmaxf(red[t], red[t + s]); __syncthreads(); }
    float zmax = red[0]; __syncthreads();

    float ls = 0.0f;
#pragma unroll
    for (int i = 0; i < 8; ++i) { float e = __expf(zv[i] - zmax); sw[t + i * 256] = e; ls += e; }
    red[t] = ls; __syncthreads();
    for (int s = 128; s > 0; s >>= 1) { if (t < s) red[t] += red[t + s]; __syncthreads(); }
    float invS = 1.0f / red[0]; __syncthreads();

#pragma unroll
    for (int i = 0; i < 8; ++i) { int n = t + i * 256; sw[n] = g * sw[n] * invS + (1.0f - g) * prev[n]; }
    __syncthreads();

    float wp[8]; float lp = 0.0f;
#pragma unroll
    for (int i = 0; i < 8; ++i) {
        int n = t + i * 256;
        float wsv = s0 * sw[(n + Nn - 1) & (Nn - 1)] + s1 * sw[n] + s2 * sw[(n + 1) & (Nn - 1)];
        float v = powf(wsv, gamma);
        wp[i] = v; lp += v;
    }
    red[t] = lp; __syncthreads();
    for (int s = 128; s > 0; s >>= 1) { if (t < s) red[t] += red[t + s]; __syncthreads(); }
    float inv = 1.0f / (red[0] + 1e-8f); __syncthreads();
#pragma unroll
    for (int i = 0; i < 8; ++i) wout[t + i * 256] = wp[i] * inv;
}

// ---------------- K7: read_vec[b,m] = sum_n w_read[b,n] * memory[b,n,m] ----------------
__global__ void k_readvec(const float* __restrict__ mem, const float* __restrict__ w_read,
                          float* __restrict__ read_vec) {
    int b = blockIdx.x;
    int t = threadIdx.x;                 // 256 threads
    int m = t & 63, seg = t >> 6;        // 4 segments over n
    const float* mb = mem + (long)b * Nn * Md;
    const float* wb = w_read + (long)b * Nn;
    float acc = 0.0f;
    for (int n = seg * 512; n < seg * 512 + 512; ++n)
        acc += wb[n] * mb[(long)n * Md + m];
    __shared__ float red[256];
    red[t] = acc; __syncthreads();
    if (t < 64) read_vec[b * Md + t] = red[t] + red[t + 64] + red[t + 128] + red[t + 192];
}

// ---------------- K8: streaming pass 2 — memory erase/add update ----------------
__global__ void k_update(const float* __restrict__ mem, const float* __restrict__ w_write,
                         const float* __restrict__ eras, const float* __restrict__ addv,
                         float* __restrict__ mem_new) {
    long i4 = (long)blockIdx.x * blockDim.x + threadIdx.x;
    if (i4 >= (long)Bsz * Nn * Md / 4) return;
    long i = i4 * 4;
    int b = (int)(i / ((long)Nn * Md));
    long r = i % ((long)Nn * Md);
    int n = (int)(r / Md);
    int m = (int)(r % Md);
    float wwv = w_write[(long)b * Nn + n];
    const float* mp = mem + i;
    const float* ep = eras + b * Md + m;
    const float* ap = addv + b * Md + m;
    float* op = mem_new + i;
#pragma unroll
    for (int j = 0; j < 4; ++j)
        op[j] = mp[j] * (1.0f - wwv * ep[j]) + wwv * ap[j];
}

// ---------------- K9: out = [h_new | read_vec] @ W_o^T + b_o ----------------
__global__ void k_out(const float* __restrict__ h, const float* __restrict__ rv,
                      const float* __restrict__ wo, const float* __restrict__ bo,
                      float* __restrict__ out) {
    int i = blockIdx.x * blockDim.x + threadIdx.x;
    if (i >= Bsz * Od) return;
    int b = i / Od, o = i % Od;
    const float* w  = wo + o * (Cs + Md);
    const float* hv = h + b * Cs;
    float acc = bo[o];
    for (int k = 0; k < Cs; ++k) acc += hv[k] * w[k];
    const float* r = rv + b * Md;
    for (int k = 0; k < Md; ++k) acc += r[k] * w[Cs + k];
    out[i] = acc;
}

extern "C" void kernel_launch(void* const* d_in, const int* in_sizes, int n_in,
                              void* d_out, int out_size, void* d_ws, size_t ws_size,
                              hipStream_t stream) {
    (void)in_sizes; (void)n_in; (void)out_size; (void)ws_size;

    const float* x      = (const float*)d_in[0];
    const float* memory = (const float*)d_in[1];
    const float* h      = (const float*)d_in[2];
    const float* c      = (const float*)d_in[3];
    const float* read_w = (const float*)d_in[4];
    const float* write_w= (const float*)d_in[5];
    const float* prevr  = (const float*)d_in[6];
    const float* W_ih   = (const float*)d_in[7];
    const float* W_hh   = (const float*)d_in[8];
    const float* b_ih   = (const float*)d_in[9];
    const float* b_hh   = (const float*)d_in[10];
    const float* W_r    = (const float*)d_in[11];
    const float* b_r    = (const float*)d_in[12];
    const float* W_w    = (const float*)d_in[13];
    const float* b_w    = (const float*)d_in[14];
    const float* W_o    = (const float*)d_in[15];
    const float* b_o    = (const float*)d_in[16];

    // outputs, concatenated flat in return order
    float* out      = (float*)d_out;
    float* mem_new  = out + Bsz * Od;
    float* h_new    = mem_new + (long)Bsz * Nn * Md;
    float* c_new    = h_new + Bsz * Cs;
    float* w_read   = c_new + Bsz * Cs;
    float* w_write  = w_read + (long)Bsz * Nn;
    float* read_vec = w_write + (long)Bsz * Nn;

    // workspace carve-out (256B aligned slabs)
    char* ws = (char*)d_ws;
    size_t off = 0;
    auto carve = [&](size_t bytes) -> char* {
        char* p = ws + off;
        off = (off + bytes + 255) & ~(size_t)255;
        return p;
    };
    __bf16* inp_bf  = (__bf16*)carve((size_t)Bsz * Kc * 2);
    __bf16* wcat_bf = (__bf16*)carve((size_t)Gt * Kc * 2);
    float*  gates   = (float*)carve((size_t)Bsz * Gt * 4);
    float*  aout    = (float*)carve((size_t)Bsz * Ao * 4);
    float*  params  = (float*)carve((size_t)Bsz * 16 * 4);
    float*  zr      = (float*)carve((size_t)Bsz * Nn * 4);
    float*  zw      = (float*)carve((size_t)Bsz * Nn * 4);
    float*  eras    = (float*)carve((size_t)Bsz * Md * 4);
    float*  addv    = (float*)carve((size_t)Bsz * Md * 4);

    // 0) bf16 packing
    k_pack_inp<<<(Bsz * Kc + 255) / 256, 256, 0, stream>>>(x, prevr, h, inp_bf);
    k_pack_w<<<(Gt * Kc + 255) / 256, 256, 0, stream>>>(W_ih, W_hh, wcat_bf);
    // 1) gates GEMM (WMMA bf16): 2048 tiles, 8 waves/block
    k_gates_wmma<<<256, 256, 0, stream>>>(inp_bf, wcat_bf, b_ih, b_hh, gates);
    // 2) LSTM
    k_lstm<<<(Bsz * Cs + 255) / 256, 256, 0, stream>>>(gates, c, h_new, c_new);
    // 3) read/write head projections
    k_addr_gemm<<<(Bsz * Ao + 255) / 256, 256, 0, stream>>>(h_new, W_r, b_r, W_w, b_w, aout);
    // 4) per-batch params
    k_params<<<Bsz, 64, 0, stream>>>(aout, params, eras, addv);
    // 5) cosine sims over memory (TDM-staged streaming pass 1)
    k_sim<<<(Bsz * Nn) / SIM_ROWS, 256, 0, stream>>>(memory, aout, params, zr, zw);
    // 6) addressing (softmax/interp/shift/sharpen)
    k_address<<<Bsz * 2, 256, 0, stream>>>(zr, zw, read_w, write_w, params, w_read, w_write);
    // 7) read vector
    k_readvec<<<Bsz, 256, 0, stream>>>(memory, w_read, read_vec);
    // 8) memory update (streaming pass 2)
    k_update<<<(int)(((long)Bsz * Nn * Md / 4 + 255) / 256), 256, 0, stream>>>(
        memory, w_write, eras, addv, mem_new);
    // 9) output projection
    k_out<<<(Bsz * Od + 255) / 256, 256, 0, stream>>>(h_new, read_vec, W_o, b_o, out);
}
